// MultiHeadSelfAttention_59167469469817
// MI455X (gfx1250) — matrix operand order repaired and checked by reading, not compiled, not run on hardware
//
#include <hip/hip_runtime.h>

// Problem constants
#define EMBD   768
#define HEADS  12
#define HDIM   64
#define BATCH  8
#define SEQ    1024
#define F3     (3 * EMBD)   // 2304

typedef __attribute__((ext_vector_type(4)))  unsigned short u16x4;
typedef __attribute__((ext_vector_type(8)))  unsigned short u16x8;
typedef __attribute__((ext_vector_type(16))) unsigned short u16x16;
typedef __attribute__((ext_vector_type(16))) __bf16         bf16x16;
typedef __attribute__((ext_vector_type(8)))  float          v8f;
typedef __attribute__((ext_vector_type(4)))  unsigned int   u32x4;
typedef __attribute__((ext_vector_type(8)))  int            i32x8;
typedef __attribute__((ext_vector_type(4)))  int            i32x4;

union V16 {
  u16x16  u;
  bf16x16 b;
  u16x8   h[2];
};

__device__ __forceinline__ unsigned short f2bf(float f) {
  // round-to-nearest-even f32 -> bf16
  unsigned u = __float_as_uint(f);
  u += 0x7FFFu + ((u >> 16) & 1u);
  return (unsigned short)(u >> 16);
}

// A fragment (16x32 bf16, M x K): lane<16 holds K {0..15}, lane>=16 holds {16..31}
__device__ __forceinline__ bf16x16 a_frag(const unsigned short* row, int k0, int klo) {
  V16 v;
  v.h[0] = *(const u16x8*)(row + k0 + 2 * klo);
  v.h[1] = *(const u16x8*)(row + k0 + 2 * klo + 8);
  return v.b;
}

// B fragment (32x16 bf16, K x N): per lane 16 consecutive K values of column (lane&15)
__device__ __forceinline__ bf16x16 b_frag(const unsigned short* p) {
  V16 v;
  v.u = *(const u16x16*)p;
  return v.b;
}

__device__ __forceinline__ v8f wmma_bf16(bf16x16 a, bf16x16 b, v8f c) {
  return __builtin_amdgcn_wmma_f32_16x16x32_bf16(false, a, false, b, (short)0, c, false, false);
}

__device__ __forceinline__ float halfmax(float t) {  // max over 16-lane half-wave
  t = fmaxf(t, __shfl_xor(t, 1, 32));
  t = fmaxf(t, __shfl_xor(t, 2, 32));
  t = fmaxf(t, __shfl_xor(t, 4, 32));
  t = fmaxf(t, __shfl_xor(t, 8, 32));
  return t;
}
__device__ __forceinline__ float halfsum(float t) {  // sum over 16-lane half-wave
  t += __shfl_xor(t, 1, 32);
  t += __shfl_xor(t, 2, 32);
  t += __shfl_xor(t, 4, 32);
  t += __shfl_xor(t, 8, 32);
  return t;
}

// ---------------------------------------------------------------------------
// Tensor Data Mover: 2-D tile (tile_k rows of tile_d contiguous bf16 elems,
// row stride stride_elems) from global -> LDS at byte offset lds_off.
// Descriptor layout per CDNA5 ISA (D# group0/group1).
// This toolchain uses the 6-arg builtin: (g0, g1, g2, g3, g4, cpol).
// ---------------------------------------------------------------------------
#if __has_builtin(__builtin_amdgcn_tensor_load_to_lds) && \
    __has_builtin(__builtin_amdgcn_s_wait_tensorcnt)
#define USE_TDM 1
__device__ __forceinline__ void tdm_load_2d_bf16(
    unsigned lds_off, const void* gaddr,
    unsigned tile_d, unsigned tile_k,
    unsigned tensor_d, unsigned tensor_k,
    unsigned long long stride_elems)
{
  unsigned long long ga = (unsigned long long)gaddr;
  u32x4 g0;
  g0[0] = 1u;                                   // count=1, user descriptor
  g0[1] = lds_off;                              // lds_addr (bytes)
  g0[2] = (unsigned)(ga & 0xFFFFFFFFu);         // global_addr[31:0]
  g0[3] = (unsigned)((ga >> 32) & 0x01FFFFFFu)  // global_addr[56:32]
        | (2u << 30);                           // type=2 ("image")
  i32x8 g1;
  g1[0] = (int)(1u << 16);                      // data_size=1 -> 2 bytes
  g1[1] = (int)((tensor_d & 0xFFFFu) << 16);    // tensor_dim0[15:0]
  g1[2] = (int)(((tensor_d >> 16) & 0xFFFFu) |  // tensor_dim0[31:16]
                ((tensor_k & 0xFFFFu) << 16));  // tensor_dim1[15:0]
  g1[3] = (int)(((tensor_k >> 16) & 0xFFFFu) |  // tensor_dim1[31:16]
                ((tile_d & 0xFFFFu) << 16));    // tile_dim0
  g1[4] = (int)(tile_k & 0xFFFFu);              // tile_dim1 (tile_dim2=0)
  g1[5] = (int)(unsigned)(stride_elems & 0xFFFFFFFFu);        // dim0_stride[31:0]
  g1[6] = (int)(unsigned)((stride_elems >> 32) & 0xFFFFu);    // dim0_stride[47:32]
  g1[7] = 0;
  i32x4 z4 = {};
  i32x8 z8 = {};
  __builtin_amdgcn_tensor_load_to_lds(g0, g1, z4, z4, z8, 0);
}
#else
#define USE_TDM 0
#endif

// ---------------------------------------------------------------------------
// f32 -> bf16 convert
// ---------------------------------------------------------------------------
__global__ void cvt_bf16(const float* __restrict__ in, unsigned short* __restrict__ out, int n) {
  int i = blockIdx.x * blockDim.x + threadIdx.x;
  if (i < n) out[i] = f2bf(in[i]);
}

// ---------------------------------------------------------------------------
// Generic GEMM: out[M][F] = A[M][K](bf16) @ W[F][K]^T(bf16) + bias[F](f32)
// Each wave computes a 16(M) x 32(F) tile; OUT_F32 selects f32 vs bf16 output.
// ---------------------------------------------------------------------------
template <int OUT_F32>
__global__ __launch_bounds__(256) void gemm_bf16(
    const unsigned short* __restrict__ A,
    const unsigned short* __restrict__ W,
    const float* __restrict__ bias,
    void* __restrict__ out,
    int M, int K, int F)
{
  const int lane = threadIdx.x & 31;
  const int wave = threadIdx.x >> 5;
  const int tilesF = F >> 5;
  const int gw = blockIdx.x * (blockDim.x >> 5) + wave;
  const int tm = gw / tilesF;
  const int tf = gw % tilesF;
  if (tm * 16 >= M) return;  // wave-uniform

  const int nsel = lane & 15;
  const int klo  = (lane < 16) ? 0 : 8;
  const int kb16 = (lane < 16) ? 0 : 16;

  const unsigned short* arow = A + (size_t)(tm * 16 + nsel) * K;
  const int f0 = tf * 32 + nsel;
  const int f1 = f0 + 16;
  const unsigned short* w0 = W + (size_t)f0 * K + kb16;
  const unsigned short* w1 = W + (size_t)f1 * K + kb16;

  v8f c0 = {}, c1 = {};
  for (int kk = 0; kk < K; kk += 32) {
    bf16x16 a = a_frag(arow, kk, klo);
    c0 = wmma_bf16(a, b_frag(w0 + kk), c0);
    c1 = wmma_bf16(a, b_frag(w1 + kk), c1);
  }

  const float bias0 = bias[f0];
  const float bias1 = bias[f1];
  const int mbase = tm * 16 + ((lane < 16) ? 0 : 8);
#pragma unroll
  for (int r = 0; r < 8; ++r) {
    size_t row = (size_t)(mbase + r);
    float v0 = c0[r] + bias0;
    float v1 = c1[r] + bias1;
    if (OUT_F32) {
      float* o = (float*)out;
      o[row * F + f0] = v0;
      o[row * F + f1] = v1;
    } else {
      unsigned short* o = (unsigned short*)out;
      o[row * F + f0] = f2bf(v0);
      o[row * F + f1] = f2bf(v1);
    }
  }
}

// ---------------------------------------------------------------------------
// Attention: per block = one (b, h, 128-row slab). 8 waves x 16 query rows.
// Three-pass softmax (max / sumexp / emit+PV) over 128-key LDS chunks.
// Dynamic LDS layout (bytes):
//   [0,      16384)  k_lds  [128 keys][64 d]
//   [16384,  34816)  vT_lds [64 d][144 keys, padded]
//   [34816,  43008)  p_lds  [8 waves][16][32]
// ---------------------------------------------------------------------------
#define LDS_BYTES 43008

__global__ __launch_bounds__(256) void attn_kernel(
    const unsigned short* __restrict__ qkv,   // [B][SEQ][2304] bf16 (q|k|v per head)
    float* __restrict__ attn_out,             // [B][H][SEQ][SEQ] f32
    unsigned short* __restrict__ ctx_out)     // [B][SEQ][EMBD] bf16
{
  extern __shared__ unsigned short smem[];
  unsigned short (*k_lds)[64]      = (unsigned short (*)[64])smem;
  unsigned short (*vT_lds)[144]    = (unsigned short (*)[144])(smem + 8192);
  unsigned short (*p_lds)[16][32]  = (unsigned short (*)[16][32])(smem + 17408);

  const int lane = threadIdx.x & 31;
  const int wave = threadIdx.x >> 5;
  const int idx  = blockIdx.x;
  const int rb = idx & 7;
  const int h  = (idx >> 3) % HEADS;
  const int b  = idx / (8 * HEADS);
  const int rowbase = rb * 128 + wave * 16;

  const int nsel = lane & 15;
  const int klo  = (lane < 16) ? 0 : 8;
  const int kb16 = (lane < 16) ? 0 : 16;
  const int mbase = (lane < 16) ? 0 : 8;

  // Q fragments for this wave's 16 rows (d = 0..63 -> two K-steps of 32)
  const unsigned short* qrow =
      qkv + ((size_t)(b * SEQ + rowbase + nsel)) * F3 + h * HDIM;
  bf16x16 qa0 = a_frag(qrow, 0,  klo);
  bf16x16 qa1 = a_frag(qrow, 32, klo);

  const float scale = 0.125f;  // 1/sqrt(64)
  const unsigned short* kbase_g = qkv + (size_t)b * SEQ * F3 + EMBD + h * HDIM;

  // ---------------- pass A: per-lane raw row max (no cross-lane ops) --------
  float m[8];
#pragma unroll
  for (int r = 0; r < 8; ++r) m[r] = -3.0e38f;

  for (int c0 = 0; c0 < SEQ; c0 += 128) {
    __syncthreads();
#if USE_TDM
    if (wave == 0) {
      tdm_load_2d_bf16(0u, kbase_g + (size_t)c0 * F3,
                       HDIM, 128, HDIM, 128, (unsigned long long)F3);
      __builtin_amdgcn_s_wait_tensorcnt(0);
    }
#else
    {
      int key = threadIdx.x >> 1;
      int db  = (threadIdx.x & 1) * 32;
      const u16x8* src = (const u16x8*)(kbase_g + (size_t)(c0 + key) * F3 + db);
      u16x8* dst = (u16x8*)&k_lds[key][db];
      dst[0] = src[0]; dst[1] = src[1]; dst[2] = src[2]; dst[3] = src[3];
    }
#endif
    __syncthreads();
    for (int j = 0; j < 128; j += 16) {
      v8f c = {};
      const unsigned short* kp = &k_lds[j + nsel][kb16];
      c = wmma_bf16(qa0, b_frag(kp),      c);
      c = wmma_bf16(qa1, b_frag(kp + 32), c);
#pragma unroll
      for (int r = 0; r < 8; ++r) m[r] = fmaxf(m[r], c[r]);
    }
  }
#pragma unroll
  for (int r = 0; r < 8; ++r) m[r] = halfmax(m[r]) * scale;  // final row max (scaled)

  // ---------------- pass B: per-lane sum of exp ----------------------------
  float l[8];
#pragma unroll
  for (int r = 0; r < 8; ++r) l[r] = 0.0f;

  for (int c0 = 0; c0 < SEQ; c0 += 128) {
    __syncthreads();
#if USE_TDM
    if (wave == 0) {
      tdm_load_2d_bf16(0u, kbase_g + (size_t)c0 * F3,
                       HDIM, 128, HDIM, 128, (unsigned long long)F3);
      __builtin_amdgcn_s_wait_tensorcnt(0);
    }
#else
    {
      int key = threadIdx.x >> 1;
      int db  = (threadIdx.x & 1) * 32;
      const u16x8* src = (const u16x8*)(kbase_g + (size_t)(c0 + key) * F3 + db);
      u16x8* dst = (u16x8*)&k_lds[key][db];
      dst[0] = src[0]; dst[1] = src[1]; dst[2] = src[2]; dst[3] = src[3];
    }
#endif
    __syncthreads();
    for (int j = 0; j < 128; j += 16) {
      v8f c = {};
      const unsigned short* kp = &k_lds[j + nsel][kb16];
      c = wmma_bf16(qa0, b_frag(kp),      c);
      c = wmma_bf16(qa1, b_frag(kp + 32), c);
#pragma unroll
      for (int r = 0; r < 8; ++r) l[r] += __expf(c[r] * scale - m[r]);
    }
  }
  float linv[8];
#pragma unroll
  for (int r = 0; r < 8; ++r) linv[r] = 1.0f / halfsum(l[r]);

  // ---------------- pass C: emit normalized probs + P @ V ------------------
  v8f ctx0 = {}, ctx1 = {}, ctx2 = {}, ctx3 = {};

  for (int c0 = 0; c0 < SEQ; c0 += 128) {
    __syncthreads();
#if USE_TDM
    if (wave == 0) {
      tdm_load_2d_bf16(0u, kbase_g + (size_t)c0 * F3,
                       HDIM, 128, HDIM, 128, (unsigned long long)F3);
    }
#else
    {
      int key = threadIdx.x >> 1;
      int db  = (threadIdx.x & 1) * 32;
      const u16x8* src = (const u16x8*)(kbase_g + (size_t)(c0 + key) * F3 + db);
      u16x8* dst = (u16x8*)&k_lds[key][db];
      dst[0] = src[0]; dst[1] = src[1]; dst[2] = src[2]; dst[3] = src[3];
    }
#endif
    {  // V chunk, transposed into vT_lds: thread = 4(d) x 8(key) sub-tile
      int d0 = (threadIdx.x & 15) * 4;
      int k0 = (threadIdx.x >> 4) * 8;
      unsigned short tmp[8][4];
#pragma unroll
      for (int k = 0; k < 8; ++k) {
        u16x4 v = *(const u16x4*)(qkv +
            ((size_t)(b * SEQ + c0 + k0 + k)) * F3 + 2 * EMBD + h * HDIM + d0);
        tmp[k][0] = v[0]; tmp[k][1] = v[1]; tmp[k][2] = v[2]; tmp[k][3] = v[3];
      }
#pragma unroll
      for (int i = 0; i < 4; ++i) {
        u16x8 row;
#pragma unroll
        for (int k = 0; k < 8; ++k) row[k] = tmp[k][i];
        *(u16x8*)&vT_lds[d0 + i][k0] = row;
      }
    }
#if USE_TDM
    if (wave == 0) __builtin_amdgcn_s_wait_tensorcnt(0);
#endif
    __syncthreads();

    for (int j2 = 0; j2 < 128; j2 += 32) {
#pragma unroll
      for (int sub = 0; sub < 2; ++sub) {
        int j = j2 + sub * 16;
        v8f c = {};
        const unsigned short* kp = &k_lds[j + nsel][kb16];
        c = wmma_bf16(qa0, b_frag(kp),      c);
        c = wmma_bf16(qa1, b_frag(kp + 32), c);
#pragma unroll
        for (int r = 0; r < 8; ++r) {
          float p = __expf(c[r] * scale - m[r]) * linv[r];
          int row = rowbase + mbase + r;
          attn_out[((size_t)(b * HEADS + h) * SEQ + row) * SEQ + c0 + j + nsel] = p;
          p_lds[wave][mbase + r][sub * 16 + nsel] = f2bf(p);
        }
      }
      asm volatile("s_wait_dscnt 0" ::: "memory");  // p_lds RAW (same wave)
      bf16x16 pa = a_frag(&p_lds[wave][nsel][0], 0, klo);
      const int kcol = j2 + kb16;
      ctx0 = wmma_bf16(pa, b_frag(&vT_lds[ 0 + nsel][kcol]), ctx0);
      ctx1 = wmma_bf16(pa, b_frag(&vT_lds[16 + nsel][kcol]), ctx1);
      ctx2 = wmma_bf16(pa, b_frag(&vT_lds[32 + nsel][kcol]), ctx2);
      ctx3 = wmma_bf16(pa, b_frag(&vT_lds[48 + nsel][kcol]), ctx3);
    }
  }

  // write ctx (bf16) at [b, row, h*64 + d]
#pragma unroll
  for (int r = 0; r < 8; ++r) {
    int row = rowbase + mbase + r;
    size_t base = ((size_t)(b * SEQ + row)) * EMBD + h * HDIM + nsel;
    ctx_out[base +  0] = f2bf(ctx0[r]);
    ctx_out[base + 16] = f2bf(ctx1[r]);
    ctx_out[base + 32] = f2bf(ctx2[r]);
    ctx_out[base + 48] = f2bf(ctx3[r]);
  }
}

// ---------------------------------------------------------------------------
extern "C" void kernel_launch(void* const* d_in, const int* in_sizes, int n_in,
                              void* d_out, int out_size, void* d_ws, size_t ws_size,
                              hipStream_t stream) {
  const float* x     = (const float*)d_in[0];   // [8,1024,768]
  const float* w_qkv = (const float*)d_in[1];   // [2304,768]
  const float* b_qkv = (const float*)d_in[2];   // [2304]
  const float* w_out = (const float*)d_in[3];   // [768,768]
  const float* b_out = (const float*)d_in[4];   // [768]

  const int n_x   = BATCH * SEQ * EMBD;   // 6291456
  const int n_wq  = F3 * EMBD;            // 1769472
  const int n_wo  = EMBD * EMBD;          // 589824
  const int n_qkv = BATCH * SEQ * F3;     // 18874368

  unsigned short* x_bf   = (unsigned short*)d_ws;
  unsigned short* wq_bf  = x_bf + n_x;
  unsigned short* wo_bf  = wq_bf + n_wq;
  unsigned short* qkv_bf = wo_bf + n_wo;
  unsigned short* ctx_bf = qkv_bf + n_qkv;   // total ws ~ 68 MB

  float* out_o  = (float*)d_out;                         // [8,1024,768]
  float* attn_o = out_o + (size_t)BATCH * SEQ * EMBD;    // [8,12,1024,1024]

  cvt_bf16<<<(n_x  + 255) / 256, 256, 0, stream>>>(x,     x_bf,  n_x);
  cvt_bf16<<<(n_wq + 255) / 256, 256, 0, stream>>>(w_qkv, wq_bf, n_wq);
  cvt_bf16<<<(n_wo + 255) / 256, 256, 0, stream>>>(w_out, wo_bf, n_wo);

  {  // QKV projection: [8192 x 2304]
    int M = BATCH * SEQ, K = EMBD, F = F3;
    int waves = (M / 16) * (F / 32);  // 36864
    gemm_bf16<0><<<waves / 8, 256, 0, stream>>>(x_bf, wq_bf, b_qkv, (void*)qkv_bf, M, K, F);
  }

  attn_kernel<<<BATCH * HEADS * 8, 256, LDS_BYTES, stream>>>(qkv_bf, attn_o, ctx_bf);

  {  // Output projection: [8192 x 768]
    int M = BATCH * SEQ, K = EMBD, F = EMBD;
    int waves = (M / 16) * (F / 32);  // 12288
    gemm_bf16<1><<<waves / 8, 256, 0, stream>>>(ctx_bf, wo_bf, b_out, (void*)out_o, M, K, F);
  }
}
